// GatedAttention_53429393162351
// MI455X (gfx1250) — matrix operand order repaired and checked by reading, not compiled, not run on hardware
//
#include <hip/hip_runtime.h>
#include <hip/hip_bf16.h>

// Gated linear attention for MI455X (gfx1250, wave32, WMMA + TDM).
// Heavy path: 5x [32768x1024]x[1024x1024] GEMMs in bf16 WMMA (f32 accum),
//             A/B tiles DMA'd into double-buffered LDS by the Tensor Data Mover.
// Light path: kv / ksum / scores einsums as bandwidth-bound VALU kernels.

#define DIM      1024
#define HEADS    16
#define HEAD_DIM 64
#define BATCH    4
#define SEQ      8192
#define BN_TOT   (BATCH * SEQ)   // 32768 rows

typedef __bf16         v16bf  __attribute__((ext_vector_type(16)));
typedef float          v8f    __attribute__((ext_vector_type(8)));
typedef unsigned int   u32x4  __attribute__((ext_vector_type(4)));
typedef unsigned int   u32x2  __attribute__((ext_vector_type(2)));
typedef unsigned short u16;

#if __has_builtin(__builtin_amdgcn_tensor_load_to_lds) && \
    __has_builtin(__builtin_amdgcn_s_wait_tensorcnt)
#define HAVE_TDM 1
#else
#define HAVE_TDM 0
#endif

__device__ __forceinline__ u16 f2bf(float f) {
    union { float f; unsigned u; } v; v.f = f;
    unsigned r = v.u + 0x7FFFu + ((v.u >> 16) & 1u);   // round-to-nearest-even
    return (u16)(r >> 16);
}
__device__ __forceinline__ float bf2f(u16 h) {
    union { unsigned u; float f; } v; v.u = ((unsigned)h) << 16;
    return v.f;
}

// ---------------------------------------------------------------- converts
__global__ __launch_bounds__(256) void cvt_x_kernel(const float* __restrict__ x,
                                                    u16* __restrict__ xb) {
    size_t base = ((size_t)blockIdx.x * 256 + threadIdx.x) * 8;
    u32x4 out;
#pragma unroll
    for (int j = 0; j < 4; ++j) {
        float a = x[base + 2 * j];
        float b = x[base + 2 * j + 1];
        out[j] = (unsigned)f2bf(a) | ((unsigned)f2bf(b) << 16);
    }
    *(u32x4*)(xb + base) = out;
}

// wT[n][k] = bf16(w[k][n]) : GEMM B-tiles then load contiguously along K
__global__ __launch_bounds__(256) void cvt_wT_kernel(const float* __restrict__ w,
                                                     u16* __restrict__ wT) {
    int gid = blockIdx.x * 256 + threadIdx.x;
    int n = gid >> 10;
    int k = gid & 1023;
    wT[(size_t)n * DIM + k] = f2bf(w[(size_t)k * DIM + n]);
}

// ---------------------------------------------------------------- WMMA GEMM
// C[M=32768, N=1024] = A[M,K=1024](bf16) * Wt[N,K](bf16)^T + bias, fused epilogue.
// Block: 256 thr (8 waves), tile 128x128x32. Wave -> 32(M) x 64(N) = 2x4 frags.
#define BM  128
#define BNT 128
#define BK  32
#define AST 40   // padded LDS row stride in u16 (80B: 16B-aligned, bank-spread,
#define BST 40   //  == TDM pad: 4 DWORDs every 16 DWORDs)

enum { EP_GATE = 0, EP_FMAP = 1, EP_LIN = 2, EP_OUT = 3 };

#if HAVE_TDM
typedef unsigned int ui32x4 __attribute__((ext_vector_type(4)));
typedef int          i32x4  __attribute__((ext_vector_type(4)));
typedef int          i32x8  __attribute__((ext_vector_type(8)));

// DMA one 2D bf16 tile (BK x rows=BM/BNT) from global (row stride DIM elements)
// into LDS, inserting 4 DWORDs of pad after every 16 DWORDs (row stride -> 80B).
__device__ __forceinline__ void tdm_load_tile(const u16* __restrict__ gtile,
                                              unsigned lds_addr,
                                              unsigned tensor_rows) {
    unsigned long long ga = (unsigned long long)(size_t)gtile;
    ui32x4 g0;
    g0[0] = 1u;                                            // count=1, user mode
    g0[1] = lds_addr;                                      // lds_addr [63:32]
    g0[2] = (unsigned)(ga & 0xFFFFFFFFu);                  // global_addr lo
    g0[3] = (unsigned)((ga >> 32) & 0x01FFFFFFu)           // global_addr hi
          | (2u << 30);                                    // type=2 ("image")
    i32x8 g1;
    g1[0] = (int)((1u << 16)        // data_size = 2 bytes
                | (1u << 20)        // pad_enable
                | (3u << 22)        // pad_interval: 16 DWORDs
                | (3u << 25));      // pad_amount:   4 DWORDs
    g1[1] = (int)(((unsigned)DIM & 0xFFFFu) << 16);        // tensor_dim0 lo16
    g1[2] = (int)(((unsigned)DIM >> 16)                    // tensor_dim0 hi16
                | ((tensor_rows & 0xFFFFu) << 16));        // tensor_dim1 lo16
    g1[3] = (int)((tensor_rows >> 16)                      // tensor_dim1 hi16
                | ((unsigned)BK << 16));                   // tile_dim0 = 32
    g1[4] = (int)(BM);                                     // tile_dim1=128, dim2=0
    g1[5] = (int)(DIM);                                    // dim0_stride lo32
    g1[6] = 0;                                             // dim0_stride hi, dim1_stride lo
    g1[7] = 0;
    i32x4 z4 = {0, 0, 0, 0};
#if __clang_major__ >= 23
    i32x8 z8 = {0, 0, 0, 0, 0, 0, 0, 0};
    __builtin_amdgcn_tensor_load_to_lds(g0, g1, z4, z4, z8, 0);
#else
    __builtin_amdgcn_tensor_load_to_lds(g0, g1, z4, z4, 0);
#endif
}

__device__ __forceinline__ unsigned lds_off(const void* p) {
    return (unsigned)(unsigned long long)(size_t)p;   // flat LDS addr[31:0] = offset
}
#endif

template <int MODE>
__global__ __launch_bounds__(256) void gemm_bf16_kernel(
    const u16* __restrict__ Ag, const u16* __restrict__ Wt,
    const float* __restrict__ bias, void* __restrict__ outp)
{
#if HAVE_TDM
    __shared__ u16 As[2][BM * AST];
    __shared__ u16 Bs[2][BNT * BST];
#else
    __shared__ u16 As[1][BM * AST];
    __shared__ u16 Bs[1][BNT * BST];
#endif

    const int tid  = threadIdx.x;
    const int m0   = blockIdx.y * BM;
    const int n0   = blockIdx.x * BNT;
    const int wave = tid >> 5;
    const int lane = tid & 31;
    const int l15  = lane & 15;
    const int hi   = (lane >> 4) & 1;
    const int wm   = wave >> 1;   // 0..3 -> M offset wm*32
    const int wn   = wave & 1;    // 0..1 -> N offset wn*64

    v8f acc[2][4];
#pragma unroll
    for (int i = 0; i < 2; ++i)
#pragma unroll
        for (int j = 0; j < 4; ++j)
#pragma unroll
            for (int r = 0; r < 8; ++r) acc[i][j][r] = 0.0f;

    const int nk = DIM / BK;   // 32 K-steps

#if HAVE_TDM
    if (tid < 32) {            // one wave drives the TDM (EXEC-independent DMA)
        tdm_load_tile(Ag + (size_t)m0 * DIM, lds_off(&As[0][0]), BN_TOT);
        tdm_load_tile(Wt + (size_t)n0 * DIM, lds_off(&Bs[0][0]), DIM);
    }
#endif

    for (int ks = 0; ks < nk; ++ks) {
#if HAVE_TDM
        if (tid < 32) {
            if (ks + 1 < nk) {   // prefetch next tiles into the other buffer
                int nb = (ks + 1) & 1;
                tdm_load_tile(Ag + (size_t)m0 * DIM + (ks + 1) * BK,
                              lds_off(&As[nb][0]), BN_TOT);
                tdm_load_tile(Wt + (size_t)n0 * DIM + (ks + 1) * BK,
                              lds_off(&Bs[nb][0]), DIM);
                __builtin_amdgcn_s_wait_tensorcnt(2);   // current tiles landed
            } else {
                __builtin_amdgcn_s_wait_tensorcnt(0);
            }
        }
        __syncthreads();                    // publish buffer ks&1
        const u16* Asb = &As[ks & 1][0];
        const u16* Bsb = &Bs[ks & 1][0];
#else
        __syncthreads();
        // Stage tiles: 512 chunks of 8 bf16 each for A and for B(=Wt row-major).
#pragma unroll
        for (int i = 0; i < 2; ++i) {
            int c   = tid + i * 256;
            int row = c >> 2;
            int c8  = (c & 3) << 3;
            *(u32x4*)&As[0][row * AST + c8] =
                *(const u32x4*)(Ag + (size_t)(m0 + row) * DIM + ks * BK + c8);
            *(u32x4*)&Bs[0][row * BST + c8] =
                *(const u32x4*)(Wt + (size_t)(n0 + row) * DIM + ks * BK + c8);
        }
        if (ks + 1 < nk) {   // hint next K-slab into cache (global_prefetch_b8)
            __builtin_prefetch(Ag + (size_t)(m0 + (tid >> 1)) * DIM + (ks + 1) * BK, 0, 1);
            __builtin_prefetch(Wt + (size_t)(n0 + (tid >> 1)) * DIM + (ks + 1) * BK, 0, 1);
        }
        __syncthreads();
        const u16* Asb = &As[0][0];
        const u16* Bsb = &Bs[0][0];
#endif

        // A frag (16x32 bf16): lane<16 row=l15 K{0..7,16..23}; lane>=16 K{8..15,24..31}
        union { v16bf v; u32x4 q[2]; } af[2], bf[4];
#pragma unroll
        for (int fm = 0; fm < 2; ++fm) {
            int r = (wm * 32 + fm * 16 + l15) * AST + (hi ? 8 : 0);
            af[fm].q[0] = *(const u32x4*)&Asb[r];
            af[fm].q[1] = *(const u32x4*)&Asb[r + 16];
        }
        // B frag (32x16 bf16): lane<16 col=l15 K{0..7,16..23}; lane>=16 K{8..15,24..31} (Bs is [N][K])
#pragma unroll
        for (int fn = 0; fn < 4; ++fn) {
            int r = (wn * 64 + fn * 16 + l15) * BST + (hi ? 8 : 0);
            bf[fn].q[0] = *(const u32x4*)&Bsb[r];
            bf[fn].q[1] = *(const u32x4*)&Bsb[r + 16];
        }
#pragma unroll
        for (int fm = 0; fm < 2; ++fm)
#pragma unroll
            for (int fn = 0; fn < 4; ++fn)
                acc[fm][fn] = __builtin_amdgcn_wmma_f32_16x16x32_bf16(
                    false, af[fm].v, false, bf[fn].v, (short)0, acc[fm][fn],
                    false, false);

#if HAVE_TDM
        __syncthreads();   // all waves done reading buffer ks&1 (TDM reuses it next)
#endif
    }

    // Epilogue: C/D layout -> VGPR r: M = r (lanes 0-15) or r+8 (lanes 16-31), N = l15.
#pragma unroll
    for (int fm = 0; fm < 2; ++fm) {
#pragma unroll
        for (int fn = 0; fn < 4; ++fn) {
            int col   = n0 + wn * 64 + fn * 16 + l15;
            int mrow  = m0 + wm * 32 + fm * 16 + (hi ? 8 : 0);
            float bv  = bias[col];
#pragma unroll
            for (int r = 0; r < 8; ++r) {
                float v  = acc[fm][fn][r] + bv;
                size_t o = (size_t)(mrow + r) * DIM + col;
                if (MODE == EP_GATE) {
                    ((u16*)outp)[o] = f2bf(1.0f / (1.0f + __expf(-v)));
                } else if (MODE == EP_FMAP) {
                    ((u16*)outp)[o] = f2bf(v > 0.0f ? v + 1.0f : __expf(v));
                } else if (MODE == EP_LIN) {
                    ((u16*)outp)[o] = f2bf(v);
                } else {
                    ((float*)outp)[o] = v;
                }
            }
        }
    }
}

// ---------------------------------------------------------------- ksum
// ksum[b][c] = sum_n k[b,n,c]   (c = h*64+d)
__global__ __launch_bounds__(256) void ksum_kernel(const u16* __restrict__ kb,
                                                   float* __restrict__ ksum) {
    int gid = blockIdx.x * 256 + threadIdx.x;   // 0..4095
    int b = gid >> 10;
    int c = gid & 1023;
    float s = 0.0f;
    for (int n = 0; n < SEQ; ++n)
        s += bf2f(kb[(size_t)(b * SEQ + n) * DIM + c]);
    ksum[gid] = s;
}

// ---------------------------------------------------------------- kv
// kv[b,h,d,e] = sum_n k[b,n,h,d] * v[b,n,h,e]. One block per (b,h).
__global__ __launch_bounds__(256) void kv_kernel(const u16* __restrict__ kb,
                                                 const u16* __restrict__ vb,
                                                 float* __restrict__ kvb) {
    const int h = blockIdx.x, b = blockIdx.y;
    const int tid = threadIdx.x;
    const int d  = tid >> 2;          // 0..63
    const int e0 = (tid & 3) * 16;    // 16 accumulators along e
    __shared__ float ks_[32][64];
    __shared__ float vs_[32][64];

    float acc[16];
#pragma unroll
    for (int e = 0; e < 16; ++e) acc[e] = 0.0f;

    for (int nc = 0; nc < SEQ; nc += 32) {
        __syncthreads();
        int row  = tid >> 3;
        int col8 = (tid & 7) * 8;
        size_t g = (size_t)(b * SEQ + nc + row) * DIM + h * HEAD_DIM + col8;
        u32x4 kr = *(const u32x4*)(kb + g);
        u32x4 vr = *(const u32x4*)(vb + g);
#pragma unroll
        for (int j = 0; j < 4; ++j) {
            ks_[row][col8 + 2 * j]     = bf2f((u16)(kr[j] & 0xFFFFu));
            ks_[row][col8 + 2 * j + 1] = bf2f((u16)(kr[j] >> 16));
            vs_[row][col8 + 2 * j]     = bf2f((u16)(vr[j] & 0xFFFFu));
            vs_[row][col8 + 2 * j + 1] = bf2f((u16)(vr[j] >> 16));
        }
        __syncthreads();
#pragma unroll 4
        for (int nn = 0; nn < 32; ++nn) {
            float kd = ks_[nn][d];
#pragma unroll
            for (int e = 0; e < 16; ++e) acc[e] += kd * vs_[nn][e0 + e];
        }
    }
    float* dst = kvb + (((size_t)(b * HEADS + h) * HEAD_DIM + d) * HEAD_DIM + e0);
#pragma unroll
    for (int e = 0; e < 16; ++e) dst[e] = acc[e];
}

// ---------------------------------------------------------------- scores
// s[n,h,e] = gate[n,h*64+e] * z[n,h] * sum_d q[n,h,d]*kv[b,h,d,e]
// z = 1/(sum_d q[n,h,d]*ksum[b,h,d] + 1e-6). Block = (b, h, 16 positions).
__global__ __launch_bounds__(256) void scores_kernel(
    const u16* __restrict__ qb, const float* __restrict__ kvb,
    const float* __restrict__ ksum, const u16* __restrict__ gate,
    u16* __restrict__ sb)
{
    const int n0 = blockIdx.x * 16, h = blockIdx.y, b = blockIdx.z;
    const int tid = threadIdx.x;
    __shared__ float kvs[64][65];
    __shared__ float qs[16][64];
    __shared__ float kss[64];
    __shared__ float zs[16];

    const float* kvsrc = kvb + (size_t)(b * HEADS + h) * HEAD_DIM * HEAD_DIM;
#pragma unroll
    for (int i = 0; i < 16; ++i) {
        int idx = tid + i * 256;
        kvs[idx >> 6][idx & 63] = kvsrc[idx];
    }
    if (tid < 64) kss[tid] = ksum[b * DIM + h * HEAD_DIM + tid];
    {
        int idx = tid * 4;
        int p = idx >> 6, d0 = idx & 63;
        u32x2 raw = *(const u32x2*)(qb + (size_t)(b * SEQ + n0 + p) * DIM +
                                    h * HEAD_DIM + d0);
        qs[p][d0 + 0] = bf2f((u16)(raw.x & 0xFFFFu));
        qs[p][d0 + 1] = bf2f((u16)(raw.x >> 16));
        qs[p][d0 + 2] = bf2f((u16)(raw.y & 0xFFFFu));
        qs[p][d0 + 3] = bf2f((u16)(raw.y >> 16));
    }
    __syncthreads();
    if (tid < 16) {
        float s = 0.0f;
#pragma unroll 8
        for (int d = 0; d < 64; ++d) s += qs[tid][d] * kss[d];
        zs[tid] = 1.0f / (s + 1e-6f);
    }
    __syncthreads();

    const int p  = tid >> 4;
    const int e0 = (tid & 15) * 4;
    float a0 = 0.f, a1 = 0.f, a2 = 0.f, a3 = 0.f;
#pragma unroll 8
    for (int d = 0; d < 64; ++d) {
        float qd = qs[p][d];
        a0 += qd * kvs[d][e0 + 0];
        a1 += qd * kvs[d][e0 + 1];
        a2 += qd * kvs[d][e0 + 2];
        a3 += qd * kvs[d][e0 + 3];
    }
    size_t row = (size_t)(b * SEQ + n0 + p) * DIM + h * HEAD_DIM + e0;
    float z = zs[p];
    float r0 = a0 * z * bf2f(gate[row + 0]);
    float r1 = a1 * z * bf2f(gate[row + 1]);
    float r2 = a2 * z * bf2f(gate[row + 2]);
    float r3 = a3 * z * bf2f(gate[row + 3]);
    u32x2 packed;
    packed.x = (unsigned)f2bf(r0) | ((unsigned)f2bf(r1) << 16);
    packed.y = (unsigned)f2bf(r2) | ((unsigned)f2bf(r3) << 16);
    *(u32x2*)(sb + row) = packed;
}

// ---------------------------------------------------------------- launch
extern "C" void kernel_launch(void* const* d_in, const int* in_sizes, int n_in,
                              void* d_out, int out_size, void* d_ws, size_t ws_size,
                              hipStream_t stream) {
    const float* x    = (const float*)d_in[0];
    const float* wg_w = (const float*)d_in[1];
    const float* wg_b = (const float*)d_in[2];
    const float* wq_w = (const float*)d_in[3];
    const float* wq_b = (const float*)d_in[4];
    const float* wk_w = (const float*)d_in[5];
    const float* wk_b = (const float*)d_in[6];
    const float* wv_w = (const float*)d_in[7];
    const float* wv_b = (const float*)d_in[8];
    const float* wo_w = (const float*)d_in[9];
    const float* wo_b = (const float*)d_in[10];

    // Workspace layout (~331 MB total)
    char* ws = (char*)d_ws;
    const size_t SZ_ACT = (size_t)BN_TOT * DIM * sizeof(u16);   // 64 MB
    u16*   xb    = (u16*)(ws + 0 * SZ_ACT);
    u16*   qb    = (u16*)(ws + 1 * SZ_ACT);
    u16*   kb    = (u16*)(ws + 2 * SZ_ACT);
    u16*   vb    = (u16*)(ws + 3 * SZ_ACT);
    u16*   gateb = (u16*)(ws + 4 * SZ_ACT);
    u16*   wTg   = (u16*)(ws + 5 * SZ_ACT);
    u16*   wTq   = wTg + (size_t)DIM * DIM;
    u16*   wTk   = wTq + (size_t)DIM * DIM;
    u16*   wTv   = wTk + (size_t)DIM * DIM;
    u16*   wTo   = wTv + (size_t)DIM * DIM;
    float* kvb   = (float*)(wTo + (size_t)DIM * DIM);
    float* ksumb = kvb + (size_t)BATCH * HEADS * HEAD_DIM * HEAD_DIM;
    u16*   sb    = xb;   // xb is dead after the 4 projection GEMMs -> reuse

    // 1) precision conversion
    cvt_x_kernel<<<dim3(BN_TOT * DIM / (256 * 8)), dim3(256), 0, stream>>>(x, xb);
    dim3 wgrid(DIM * DIM / 256);
    cvt_wT_kernel<<<wgrid, 256, 0, stream>>>(wg_w, wTg);
    cvt_wT_kernel<<<wgrid, 256, 0, stream>>>(wq_w, wTq);
    cvt_wT_kernel<<<wgrid, 256, 0, stream>>>(wk_w, wTk);
    cvt_wT_kernel<<<wgrid, 256, 0, stream>>>(wv_w, wTv);
    cvt_wT_kernel<<<wgrid, 256, 0, stream>>>(wo_w, wTo);

    // 2) projections (WMMA GEMMs with fused activations)
    dim3 ggrid(DIM / BNT, BN_TOT / BM);   // (8, 256)
    gemm_bf16_kernel<EP_GATE><<<ggrid, 256, 0, stream>>>(xb, wTg, wg_b, gateb);
    gemm_bf16_kernel<EP_FMAP><<<ggrid, 256, 0, stream>>>(xb, wTq, wq_b, qb);
    gemm_bf16_kernel<EP_FMAP><<<ggrid, 256, 0, stream>>>(xb, wTk, wk_b, kb);
    gemm_bf16_kernel<EP_LIN ><<<ggrid, 256, 0, stream>>>(xb, wTv, wv_b, vb);

    // 3) linear-attention state
    ksum_kernel<<<dim3(BATCH * DIM / 256), 256, 0, stream>>>(kb, ksumb);
    kv_kernel<<<dim3(HEADS, BATCH), 256, 0, stream>>>(kb, vb, kvb);
    scores_kernel<<<dim3(SEQ / 16, HEADS, BATCH), 256, 0, stream>>>(
        qb, kvb, ksumb, gateb, sb);

    // 4) output projection -> f32 d_out
    gemm_bf16_kernel<EP_OUT><<<ggrid, 256, 0, stream>>>(sb, wTo, wo_b, (float*)d_out);
}
